// SpatialAttention_50843822850476
// MI455X (gfx1250) — matrix operand order repaired and checked by reading, not compiled, not run on hardware
//
#include <hip/hip_runtime.h>
#include <hip/hip_bf16.h>
#include <cstdint>
#include <cstddef>

// Problem sizes (fixed by the reference)
#define BSZ 32
#define CCH 256
#define SEQL 32
#define HWN 4096   // 64*64

typedef __bf16 bf16_t;
typedef __attribute__((ext_vector_type(16))) __bf16 v16bf;
typedef __attribute__((ext_vector_type(8)))  float  v8f;

// ---------------------------------------------------------------------------
// Kernel 1: ctx[b] = context[b] (32x256) @ W_word (256x256), written as bf16 in
// two layouts: ctxB[b][s][c] (B-operand of attn GEMM) and ctxTB[b][c][s]
// (B-operand of the weighted-context GEMM). Tiny kernel (0.13 GFLOP total).
// ---------------------------------------------------------------------------
__global__ __launch_bounds__(256) void ctx_proj_kernel(
    const float* __restrict__ context, const float* __restrict__ Wword,
    bf16_t* __restrict__ ctxB, bf16_t* __restrict__ ctxTB) {
  __shared__ float ctxRow[SEQL * CCH];  // 32 KB: context[b] staged in LDS
  const int b = blockIdx.x;
  const float* src = context + (size_t)b * SEQL * CCH;
  for (int i = threadIdx.x; i < SEQL * CCH; i += 256) ctxRow[i] = src[i];
  __syncthreads();

  const int c = threadIdx.x;  // one output column per thread
  float acc[SEQL];
#pragma unroll
  for (int s = 0; s < SEQL; ++s) acc[s] = 0.f;
  for (int d = 0; d < CCH; ++d) {
    const float wv = Wword[d * CCH + c];  // coalesced across lanes
#pragma unroll
    for (int s = 0; s < SEQL; ++s) acc[s] = fmaf(ctxRow[s * CCH + d], wv, acc[s]);
  }
#pragma unroll
  for (int s = 0; s < SEQL; ++s) {
    const bf16_t v = (bf16_t)acc[s];
    ctxB[((size_t)b * SEQL + s) * CCH + c] = v;
    ctxTB[((size_t)b * CCH + c) * SEQL + s] = v;
  }
}

// ---------------------------------------------------------------------------
// Kernel 2: fused attn GEMM -> masked softmax -> weighted-context GEMM.
// 256 threads = 8 waves; each wave owns 16 rows of hw. Grid = 32 batches x
// 32 row-tiles. All matrix math on v_wmma_f32_16x16x32_bf16.
// ---------------------------------------------------------------------------
__global__ __launch_bounds__(256) void spatial_attn_kernel(
    const float* __restrict__ x, const unsigned char* __restrict__ mask,
    const bf16_t* __restrict__ ctxB, const bf16_t* __restrict__ ctxTB,
    float* __restrict__ outW, float* __restrict__ outA) {
  __shared__ float attn_lds[8][16 * 33];   // per-wave 16x32 tile, padded
  __shared__ unsigned char mask_sh[SEQL];

  const int b       = blockIdx.x >> 5;
  const int tile    = blockIdx.x & 31;
  const int wave    = threadIdx.x >> 5;
  const int lane    = threadIdx.x & 31;
  const int laneRow = lane & 15;
  const int hiHalf  = (lane >> 4) & 1;
  const int rowBase = tile * 128 + wave * 16;

  if (threadIdx.x < SEQL) mask_sh[threadIdx.x] = mask[b * SEQL + threadIdx.x];

  // ---------------- Step B: attn = x_tile (16x256) @ ctx^T (256x32) --------
  const float* xrow = x + ((size_t)b * HWN + rowBase + laneRow) * CCH;
  v8f acc0 = {};  // columns s = 0..15
  v8f acc1 = {};  // columns s = 16..31
#pragma unroll
  for (int k = 0; k < 8; ++k) {
    // A fragment: lanes 0-15: K=0..15, lanes 16-31:
    // K=16..31 of this 32-wide k-block; convert f32 -> bf16 on the fly.
    const float* xr = xrow + 32 * k + (hiHalf ? 16 : 0);
    const float4 a0 = *(const float4*)(xr + 0);
    const float4 a1 = *(const float4*)(xr + 4);
    const float4 a2 = *(const float4*)(xr + 8);
    const float4 a3 = *(const float4*)(xr + 12);
    v16bf A;
    A[0]  = (bf16_t)a0.x; A[1]  = (bf16_t)a0.y; A[2]  = (bf16_t)a0.z; A[3]  = (bf16_t)a0.w;
    A[4]  = (bf16_t)a1.x; A[5]  = (bf16_t)a1.y; A[6]  = (bf16_t)a1.z; A[7]  = (bf16_t)a1.w;
    A[8]  = (bf16_t)a2.x; A[9]  = (bf16_t)a2.y; A[10] = (bf16_t)a2.z; A[11] = (bf16_t)a2.w;
    A[12] = (bf16_t)a3.x; A[13] = (bf16_t)a3.y; A[14] = (bf16_t)a3.z; A[15] = (bf16_t)a3.w;

    // B fragments: B[k][n] = ctx[s=n][c]; lane = column, 16 contiguous bf16.
    const int cOff = 32 * k + (hiHalf ? 16 : 0);
    const v16bf B0 = *(const v16bf*)(ctxB + ((size_t)b * SEQL + laneRow) * CCH + cOff);
    const v16bf B1 = *(const v16bf*)(ctxB + ((size_t)b * SEQL + 16 + laneRow) * CCH + cOff);

    acc0 = __builtin_amdgcn_wmma_f32_16x16x32_bf16(false, A, false, B0, (short)0, acc0, false, false);
    acc1 = __builtin_amdgcn_wmma_f32_16x16x32_bf16(false, A, false, B1, (short)0, acc1, false, false);
  }

  // ---------------- spill D fragments to LDS in (m, s) order ---------------
  float* tl = attn_lds[wave];
  const int m0 = hiHalf * 8;
#pragma unroll
  for (int r = 0; r < 8; ++r) {
    tl[(m0 + r) * 33 + laneRow]      = acc0[r];
    tl[(m0 + r) * 33 + laneRow + 16] = acc1[r];
  }
  __syncthreads();

  // ---------------- masked softmax over s (lanes 0-15: one row each) -------
  if (lane < 16) {
    float* myrow = tl + laneRow * 33;
    float v[SEQL];
    float mx = -1e30f;
#pragma unroll
    for (int s = 0; s < SEQL; ++s) {
      float t = myrow[s];
      if (mask_sh[s]) t = -1e30f;   // masked position -> effectively -inf
      v[s] = t;
      mx = fmaxf(mx, t);
    }
    float sum = 0.f;
#pragma unroll
    for (int s = 0; s < SEQL; ++s) { const float e = __expf(v[s] - mx); v[s] = e; sum += e; }
    const float inv = 1.0f / sum;   // rows never fully masked (per reference)
    float* oa = outA + ((size_t)b * HWN + rowBase + laneRow) * SEQL;
#pragma unroll
    for (int s = 0; s < SEQL; s += 4) {
      float4 o;
      o.x = v[s + 0] * inv; o.y = v[s + 1] * inv;
      o.z = v[s + 2] * inv; o.w = v[s + 3] * inv;
      myrow[s + 0] = o.x; myrow[s + 1] = o.y; myrow[s + 2] = o.z; myrow[s + 3] = o.w;
      *(float4*)(oa + s) = o;       // word_attn output, coalesced 128b stores
    }
  }
  __syncthreads();

  // ---------------- Step C: weighted = probs (16x32) @ ctx (32x256) --------
  // A fragment rebuilt from LDS probs, contiguous K = s (16 per lane half).
  const int sBase = hiHalf ? 16 : 0;
  const float* prow = tl + laneRow * 33;
  v16bf Ap;
#pragma unroll
  for (int i = 0; i < 8; ++i) {
    Ap[i]     = (bf16_t)prow[sBase + i];
    Ap[8 + i] = (bf16_t)prow[sBase + 8 + i];
  }
  const int sOff = hiHalf ? 16 : 0;
  const bf16_t* ctxTb = ctxTB + (size_t)b * CCH * SEQL;
  float* ow = outW + ((size_t)b * HWN + rowBase) * CCH;
#pragma unroll
  for (int j = 0; j < 16; ++j) {
    // B[k][n] = ctx[s=k][c=16j+n] = ctxT[c][s]; lane = column, contiguous s.
    const v16bf Bc = *(const v16bf*)(ctxTb + (size_t)(j * 16 + laneRow) * SEQL + sOff);
    v8f accC = {};
    accC = __builtin_amdgcn_wmma_f32_16x16x32_bf16(false, Ap, false, Bc, (short)0, accC, false, false);
#pragma unroll
    for (int r = 0; r < 8; ++r)
      ow[(size_t)(m0 + r) * CCH + j * 16 + laneRow] = accC[r];
  }
}

// ---------------------------------------------------------------------------
extern "C" void kernel_launch(void* const* d_in, const int* in_sizes, int n_in,
                              void* d_out, int out_size, void* d_ws, size_t ws_size,
                              hipStream_t stream) {
  (void)in_sizes; (void)n_in; (void)out_size; (void)ws_size;
  const float*         x       = (const float*)d_in[0];
  /* d_in[1] = sentence (unused by the reference call) */
  const float*         context = (const float*)d_in[2];
  const unsigned char* mask    = (const unsigned char*)d_in[3];  // bool array
  const float*         Wword   = (const float*)d_in[4];

  bf16_t* ctxB  = (bf16_t*)d_ws;                                 // [b][s][c]
  bf16_t* ctxTB = ctxB + (size_t)BSZ * SEQL * CCH;               // [b][c][s]

  float* outW = (float*)d_out;                                   // [32,64,64,256]
  float* outA = outW + (size_t)BSZ * HWN * CCH;                  // [32,64,64,32]

  ctx_proj_kernel<<<dim3(BSZ), dim3(256), 0, stream>>>(context, Wword, ctxB, ctxTB);
  spatial_attn_kernel<<<dim3(BSZ * 32), dim3(256), 0, stream>>>(
      x, mask, ctxB, ctxTB, outW, outA);
}
